// MNIST_Model_2_levels_w_Att_10565619549010
// MI455X (gfx1250) — matrix operand order repaired and checked by reading, not compiled, not run on hardware
//
#include <hip/hip_runtime.h>
#include <hip/hip_bf16.h>
#include <cstddef>

// ---------------------------------------------------------------------------
// MIL MNIST 2-level attention model for MI455X (gfx1250, wave32, WMMA).
// conv2 (36.5 GFLOP, dominant) and att1 (3.4 GFLOP) run as f16 WMMA implicit
// GEMMs with f32 accumulation; everything else is bandwidth-trivial VALU work.
// ---------------------------------------------------------------------------

typedef _Float16 v16h __attribute__((ext_vector_type(16)));
typedef _Float16 v8h  __attribute__((ext_vector_type(8)));
typedef float    v8f  __attribute__((ext_vector_type(8)));

union Frag16 { v16h v; v8h h[2]; };

// Build a v16h A/B fragment from two 16-byte contiguous chunks.
__device__ __forceinline__ v16h load_frag(const _Float16* p0, const _Float16* p1) {
    Frag16 f;
    f.h[0] = *(const v8h*)p0;
    f.h[1] = *(const v8h*)p1;
    return f.v;
}

#define NINST   8192
#define NBAGS   128
#define FEAT    1600
#define H1SZ    (13*13*32)   // 5408 f16 per image
#define K2      288          // conv2 GEMM depth = 9*32

// ---------------------------------------------------------------------------
// Kernel 0: weight conversion/transposition to f16 + zero the exp-sum scalar.
//   w2T[co*288 + k]  = conv2_w[k*64 + co]        (k = (ky*3+kx)*32 + ci, HWIO)
//   w1T[n*1600 + k]  = att1_w1[k*128 + n]
// ---------------------------------------------------------------------------
__global__ void k_convert(const float* __restrict__ conv2_w,
                          const float* __restrict__ att1_w1,
                          _Float16* __restrict__ w2T,
                          _Float16* __restrict__ w1T,
                          float* __restrict__ esum) {
    const int i = blockIdx.x * blockDim.x + threadIdx.x;
    if (i == 0) *esum = 0.f;
    if (i < 64 * K2) {
        const int co = i / K2, k = i % K2;
        w2T[i] = (_Float16)conv2_w[k * 64 + co];
    }
    if (i < 128 * FEAT) {
        const int n = i / FEAT, k = i % FEAT;
        w1T[i] = (_Float16)att1_w1[k * 128 + n];
    }
}

// ---------------------------------------------------------------------------
// Kernel 1: conv1 + relu + maxpool2 -> h1 [N,13,13,32] as f16 (c innermost).
// One workgroup per image; image + weights staged in LDS.
// ---------------------------------------------------------------------------
__global__ void k_conv1(const float* __restrict__ x,
                        const float* __restrict__ w,
                        const float* __restrict__ b,
                        _Float16* __restrict__ h1) {
    __shared__ float sx[784];
    __shared__ float sw[288];
    __shared__ float sb[32];
    const int n = blockIdx.x, tid = threadIdx.x;
    for (int i = tid; i < 784; i += 256) sx[i] = x[(size_t)n * 784 + i];
    for (int i = tid; i < 288; i += 256) sw[i] = w[i];
    if (tid < 32) sb[tid] = b[tid];
    __syncthreads();

    _Float16* out = h1 + (size_t)n * H1SZ;
    for (int i = tid; i < H1SZ; i += 256) {
        const int c = i & 31, px = (i >> 5) % 13, py = i / (32 * 13);
        float m = -1e30f;
        #pragma unroll
        for (int dy = 0; dy < 2; ++dy)
        #pragma unroll
        for (int dx = 0; dx < 2; ++dx) {
            const int cy = 2 * py + dy, cx = 2 * px + dx;  // conv output coords (<=25)
            float s = sb[c];
            #pragma unroll
            for (int ky = 0; ky < 3; ++ky)
            #pragma unroll
            for (int kx = 0; kx < 3; ++kx)
                s += sx[(cy + ky) * 28 + (cx + kx)] * sw[(ky * 3 + kx) * 32 + c];
            m = fmaxf(m, fmaxf(s, 0.f));
        }
        out[i] = (_Float16)m;
    }
}

// ---------------------------------------------------------------------------
// Kernel 2: conv2 as implicit GEMM on WMMA + relu + maxpool2 -> emb [N,1600].
// One workgroup (8 waves) per image. Per-position rows M (121, padded to 128),
// K = 288 (9 k-blocks of 32 = the 9 filter taps, ci innermost so fragments are
// contiguous 16B chunks), Nout = 64 channels. Wave w owns M-tile w, loops the
// 4 channel tiles -> 36 v_wmma_f32_16x16x32_f16 per wave.
// A-row gather is BRANCHLESS: positions >=121 clamp to 120 (their C rows are
// discarded at the store), so no exec-mask juggling around the ds_load_b128s.
// ---------------------------------------------------------------------------
__global__ void k_conv2(const _Float16* __restrict__ h1,
                        const _Float16* __restrict__ w2T,   // [64][288] f16
                        const float* __restrict__ b2,
                        float* __restrict__ emb32,
                        _Float16* __restrict__ emb16) {
    __shared__ __align__(16) _Float16 sA[H1SZ];       // 10816 B
    __shared__ __align__(16) float    sC[121 * 64];   // 30976 B  (tot < 64KB)

    const int n = blockIdx.x, tid = threadIdx.x;

    // Stage the image's h1 tile into LDS as 16B chunks.
    {
        const uint4* g = (const uint4*)(h1 + (size_t)n * H1SZ);
        uint4* l = (uint4*)sA;
        for (int i = tid; i < H1SZ / 8; i += 256) l[i] = g[i];
    }
    __syncthreads();

    const int wave = tid >> 5;       // M-tile 0..7
    const int lane = tid & 31;
    const int half = lane >> 4;      // 0: lanes 0-15, 1: lanes 16-31
    const int nn   = lane & 15;

    // Loop-invariant clamped A-row base for this lane.
    const int p0 = 16 * wave + nn;
    const int pc = (p0 < 121) ? p0 : 120;
    const int oy = pc / 11, ox = pc % 11;
    const _Float16* abase = sA + (oy * 13 + ox) * 32 + 8 * half;

    v8f acc[4] = {};
    #pragma unroll
    for (int kb = 0; kb < 9; ++kb) {                  // filter tap (ky,kx)
        const int ky = kb / 3, kx = kb % 3;
        // A fragment: row M = position p, K-within-block = ci.
        // v16h elem j -> ci = 16*(j/8) + 8*half + (j%8): two contiguous 16B loads.
        const _Float16* ap = abase + (ky * 13 + kx) * 32;
        v16h a = load_frag(ap, ap + 16);
        #pragma unroll
        for (int nt = 0; nt < 4; ++nt) {
            // B fragment: col = nn, K = kb*32 + 16*(j/8) + 8*half + (j%8): two contiguous 16B loads.
            const int co = nt * 16 + nn;
            const _Float16* bb = w2T + (size_t)co * K2 + kb * 32 + 8 * half;
            v16h bv = load_frag(bb, bb + 16);
            acc[nt] = __builtin_amdgcn_wmma_f32_16x16x32_f16(
                false, a, false, bv, (short)0, acc[nt], false, false);
        }
    }

    // bias + relu, scatter C (row M = 8*half + r) into the LDS conv-out tile
    #pragma unroll
    for (int nt = 0; nt < 4; ++nt) {
        const int co = nt * 16 + nn;
        const float bias = b2[co];
        #pragma unroll
        for (int r = 0; r < 8; ++r) {
            const int p = 16 * wave + 8 * half + r;
            if (p < 121) sC[p * 64 + co] = fmaxf(acc[nt][r] + bias, 0.f);
        }
    }
    __syncthreads();

    // 2x2 maxpool over the 11x11 grid (rows/cols 0..9) -> [5,5,64] flat = 1600
    float*    e32 = emb32 + (size_t)n * FEAT;
    _Float16* e16 = emb16 + (size_t)n * FEAT;
    for (int i = tid; i < FEAT; i += 256) {
        const int c = i & 63, px = (i >> 6) % 5, py = i / 320;
        const int base = ((2 * py) * 11 + 2 * px) * 64 + c;
        float v = fmaxf(fmaxf(sC[base], sC[base + 64]),
                        fmaxf(sC[base + 11 * 64], sC[base + 11 * 64 + 64]));
        e32[i] = v;
        e16[i] = (_Float16)v;
    }
}

// ---------------------------------------------------------------------------
// Kernel 3: H = tanh(emb16 @ att1_w1 + b1)  [8192,128] via WMMA.
// Register-tiled 2x2: each wave owns a 32x32 output block (4 f32 accumulators,
// 200 WMMAs/wave), A fragments shared across the two column tiles and B across
// the two row tiles -> 2 b128 loads per WMMA. Grid = 128 blocks x 8 waves
// covering (8192/32)x(128/32) = 1024 wave-tiles. K = 1600 = 50 k-blocks.
// ---------------------------------------------------------------------------
__global__ void k_att1(const _Float16* __restrict__ emb16,
                       const _Float16* __restrict__ w1T,
                       const float* __restrict__ b1,
                       float* __restrict__ H) {
    const int wave = threadIdx.x >> 5;
    const int lane = threadIdx.x & 31;
    const int half = lane >> 4;
    const int nn   = lane & 15;

    const int t  = blockIdx.x * 8 + wave;   // 0..1023
    const int m2 = t >> 2;                  // 32-row block 0..255
    const int n2 = t & 3;                   // 32-col block 0..3

    const _Float16* arow0 = emb16 + (size_t)(32 * m2 + nn)      * FEAT;
    const _Float16* arow1 = emb16 + (size_t)(32 * m2 + 16 + nn) * FEAT;
    const _Float16* bcol0 = w1T   + (size_t)(32 * n2 + nn)      * FEAT;
    const _Float16* bcol1 = w1T   + (size_t)(32 * n2 + 16 + nn) * FEAT;

    v8f c00 = {}, c01 = {}, c10 = {}, c11 = {};
    for (int kb = 0; kb < 50; ++kb) {
        const int ka = kb * 32 + 8 * half;   // A fragment K offset
        const int kB = kb * 32 + 8 * half;   // B fragment K offset
        // Prefetch the A rows a few k-blocks ahead (global_prefetch_b8).
        __builtin_prefetch(arow0 + ka + 128, 0, 0);
        __builtin_prefetch(arow1 + ka + 128, 0, 0);
        v16h a0 = load_frag(arow0 + ka, arow0 + ka + 16);
        v16h a1 = load_frag(arow1 + ka, arow1 + ka + 16);
        v16h b0 = load_frag(bcol0 + kB, bcol0 + kB + 16);
        v16h b1v = load_frag(bcol1 + kB, bcol1 + kB + 16);
        c00 = __builtin_amdgcn_wmma_f32_16x16x32_f16(false, a0, false, b0,  (short)0, c00, false, false);
        c01 = __builtin_amdgcn_wmma_f32_16x16x32_f16(false, a0, false, b1v, (short)0, c01, false, false);
        c10 = __builtin_amdgcn_wmma_f32_16x16x32_f16(false, a1, false, b0,  (short)0, c10, false, false);
        c11 = __builtin_amdgcn_wmma_f32_16x16x32_f16(false, a1, false, b1v, (short)0, c11, false, false);
    }

    const int row0 = 32 * m2 + 8 * half;        // + r      (tile rows 0..15)
    const int row1 = 32 * m2 + 16 + 8 * half;   // + r      (tile rows 16..31)
    const int col0 = 32 * n2 + nn;
    const int col1 = 32 * n2 + 16 + nn;
    const float bias0 = b1[col0];
    const float bias1 = b1[col1];
    #pragma unroll
    for (int r = 0; r < 8; ++r) {
        H[(size_t)(row0 + r) * 128 + col0] = tanhf(c00[r] + bias0);
        H[(size_t)(row0 + r) * 128 + col1] = tanhf(c01[r] + bias1);
        H[(size_t)(row1 + r) * 128 + col0] = tanhf(c10[r] + bias0);
        H[(size_t)(row1 + r) * 128 + col1] = tanhf(c11[r] + bias1);
    }
}

// ---------------------------------------------------------------------------
// Kernel 4: a1 = sigmoid(H @ att1_w2 + b2); e = exp(a1); esum += sum(e).
// sigmoid in (0,1) -> exp is safe without max-subtraction.
// ---------------------------------------------------------------------------
__global__ void k_a1exp(const float* __restrict__ H,
                        const float* __restrict__ w2,
                        const float* __restrict__ b2,
                        float* __restrict__ e,
                        float* __restrict__ esum) {
    __shared__ float red[256];
    const int i = blockIdx.x * 256 + threadIdx.x;   // < 8192
    const float* h = H + (size_t)i * 128;
    float s = b2[0];
    for (int j = 0; j < 128; ++j) s += h[j] * w2[j];
    const float sig = 1.f / (1.f + expf(-s));
    const float ev = expf(sig);
    e[i] = ev;
    red[threadIdx.x] = ev;
    __syncthreads();
    for (int st = 128; st > 0; st >>= 1) {
        if (threadIdx.x < st) red[threadIdx.x] += red[threadIdx.x + st];
        __syncthreads();
    }
    if (threadIdx.x == 0) atomicAdd(esum, red[0]);
}

// ---------------------------------------------------------------------------
// Kernel 5: segment_sum( (e/esum) * emb32, first_lab ) -> second [128,1600].
// first_lab is sorted: one block per bag binary-searches its [lo,hi) range,
// so no global atomics are needed and `second` is fully rewritten each call.
// ---------------------------------------------------------------------------
__global__ void k_segsum(const int* __restrict__ lab,
                         const float* __restrict__ e,
                         const float* __restrict__ esum,
                         const float* __restrict__ emb32,
                         float* __restrict__ second) {
    __shared__ int sLo, sHi;
    const int b = blockIdx.x;
    if (threadIdx.x == 0) {
        int lo = 0, hi = NINST;
        while (lo < hi) { int m = (lo + hi) >> 1; if (lab[m] < b) lo = m + 1; else hi = m; }
        sLo = lo;
        lo = 0; hi = NINST;
        while (lo < hi) { int m = (lo + hi) >> 1; if (lab[m] < b + 1) lo = m + 1; else hi = m; }
        sHi = lo;
    }
    __syncthreads();
    const float inv = 1.f / *esum;
    float acc[7] = {0.f, 0.f, 0.f, 0.f, 0.f, 0.f, 0.f};
    for (int r = sLo; r < sHi; ++r) {
        const float wgt = e[r] * inv;
        const float* row = emb32 + (size_t)r * FEAT;
        int k = 0;
        for (int c = threadIdx.x; c < FEAT; c += 256, ++k) acc[k] += wgt * row[c];
    }
    int k = 0;
    for (int c = threadIdx.x; c < FEAT; c += 256, ++k)
        second[(size_t)b * FEAT + c] = acc[k];
}

// ---------------------------------------------------------------------------
// Kernel 6: a2[b] = sigmoid(tanh(second[b] @ att2_w1 + b1) @ att2_w2 + b2).
// 128 blocks x 128 threads (thread j = hidden unit), LDS tree reduction.
// ---------------------------------------------------------------------------
__global__ void k_att2(const float* __restrict__ second,
                       const float* __restrict__ w1,
                       const float* __restrict__ b1,
                       const float* __restrict__ w2,
                       const float* __restrict__ b2,
                       float* __restrict__ a2) {
    __shared__ float red[128];
    const int b = blockIdx.x, j = threadIdx.x;
    const float* row = second + (size_t)b * FEAT;
    float s = b1[j];
    for (int c = 0; c < FEAT; ++c) s += row[c] * w1[c * 128 + j];
    red[j] = tanhf(s) * w2[j];
    __syncthreads();
    for (int st = 64; st > 0; st >>= 1) {
        if (j < st) red[j] += red[j + st];
        __syncthreads();
    }
    if (j == 0) a2[b] = 1.f / (1.f + expf(-(red[0] + b2[0])));
}

// Kernel 7: snd_att = softmax(a2) over 128 bags (single block).
__global__ void k_snd(const float* __restrict__ a2, float* __restrict__ snd) {
    __shared__ float red[128];
    __shared__ float tot;
    const int j = threadIdx.x;
    const float v = expf(a2[j]);   // a2 in (0,1): exp is safe
    red[j] = v;
    __syncthreads();
    for (int st = 64; st > 0; st >>= 1) {
        if (j < st) red[j] += red[j + st];
        __syncthreads();
    }
    if (j == 0) tot = red[0];
    __syncthreads();
    snd[j] = v / tot;
}

// Kernel 8: outer[c] = sum_b snd[b] * second[b][c].
__global__ void k_outer(const float* __restrict__ snd,
                        const float* __restrict__ second,
                        float* __restrict__ outer) {
    const int c = blockIdx.x * 256 + threadIdx.x;
    if (c >= FEAT) return;
    float s = 0.f;
    for (int b = 0; b < NBAGS; ++b) s += snd[b] * second[(size_t)b * FEAT + c];
    outer[c] = s;
}

// Kernel 9: pred = sigmoid((outer @ cls_w1 + b1) @ cls_w2 + b2) -> d_out[0].
__global__ void k_cls(const float* __restrict__ outer,
                      const float* __restrict__ w1,
                      const float* __restrict__ b1,
                      const float* __restrict__ w2,
                      const float* __restrict__ b2,
                      float* __restrict__ out) {
    __shared__ float red[128];
    const int j = threadIdx.x;
    float s = b1[j];
    for (int c = 0; c < FEAT; ++c) s += outer[c] * w1[c * 128 + j];
    red[j] = s * w2[j];
    __syncthreads();
    for (int st = 64; st > 0; st >>= 1) {
        if (j < st) red[j] += red[j + st];
        __syncthreads();
    }
    if (j == 0) out[0] = 1.f / (1.f + expf(-(red[0] + b2[0])));
}

// ---------------------------------------------------------------------------
extern "C" void kernel_launch(void* const* d_in, const int* in_sizes, int n_in,
                              void* d_out, int out_size, void* d_ws, size_t ws_size,
                              hipStream_t stream) {
    const float* x    = (const float*)d_in[0];
    const int*   lab  = (const int*)  d_in[1];
    const float* c1w  = (const float*)d_in[2];
    const float* c1b  = (const float*)d_in[3];
    const float* c2w  = (const float*)d_in[4];
    const float* c2b  = (const float*)d_in[5];
    const float* a1w1 = (const float*)d_in[6];
    const float* a1b1 = (const float*)d_in[7];
    const float* a1w2 = (const float*)d_in[8];
    const float* a1b2 = (const float*)d_in[9];
    const float* a2w1 = (const float*)d_in[10];
    const float* a2b1 = (const float*)d_in[11];
    const float* a2w2 = (const float*)d_in[12];
    const float* a2b2 = (const float*)d_in[13];
    const float* cw1  = (const float*)d_in[14];
    const float* cb1  = (const float*)d_in[15];
    const float* cw2  = (const float*)d_in[16];
    const float* cb2  = (const float*)d_in[17];
    float* out = (float*)d_out;

    // Workspace layout (256B-aligned slabs), ~173 MB total.
    char* base = (char*)d_ws;
    size_t off = 0;
    auto alloc = [&](size_t bytes) -> void* {
        off = (off + 255) & ~(size_t)255;
        void* p = base + off;
        off += bytes;
        return p;
    };
    _Float16* h1     = (_Float16*)alloc((size_t)NINST * H1SZ * 2);     // 88.6 MB
    float*    emb32  = (float*)   alloc((size_t)NINST * FEAT * 4);     // 52.4 MB
    _Float16* emb16  = (_Float16*)alloc((size_t)NINST * FEAT * 2);     // 26.2 MB
    _Float16* w2T    = (_Float16*)alloc((size_t)64 * K2 * 2);
    _Float16* w1T    = (_Float16*)alloc((size_t)128 * FEAT * 2);
    float*    Hbuf   = (float*)   alloc((size_t)NINST * 128 * 4);      // 4.2 MB
    float*    ebuf   = (float*)   alloc((size_t)NINST * 4);
    float*    esum   = (float*)   alloc(256);
    float*    second = (float*)   alloc((size_t)NBAGS * FEAT * 4);
    float*    a2buf  = (float*)   alloc((size_t)NBAGS * 4);
    float*    snd    = (float*)   alloc((size_t)NBAGS * 4);
    float*    outer  = (float*)   alloc((size_t)FEAT * 4);

    k_convert<<<(128 * FEAT + 255) / 256, 256, 0, stream>>>(c2w, a1w1, w2T, w1T, esum);
    k_conv1  <<<NINST, 256, 0, stream>>>(x, c1w, c1b, h1);
    k_conv2  <<<NINST, 256, 0, stream>>>(h1, w2T, c2b, emb32, emb16);
    k_att1   <<<128, 256, 0, stream>>>(emb16, w1T, a1b1, Hbuf);
    k_a1exp  <<<NINST / 256, 256, 0, stream>>>(Hbuf, a1w2, a1b2, ebuf, esum);
    k_segsum <<<NBAGS, 256, 0, stream>>>(lab, ebuf, esum, emb32, second);
    k_att2   <<<NBAGS, 128, 0, stream>>>(second, a2w1, a2b1, a2w2, a2b2, a2buf);
    k_snd    <<<1, 128, 0, stream>>>(a2buf, snd);
    k_outer  <<<(FEAT + 255) / 256, 256, 0, stream>>>(snd, second, outer);
    k_cls    <<<1, 128, 0, stream>>>(outer, cw1, cb1, cw2, cb2, out);
}
